// TopKAttention_26396869001821
// MI455X (gfx1250) — matrix operand order repaired and checked by reading, not compiled, not run on hardware
//
#include <hip/hip_runtime.h>
#include <hip/hip_bf16.h>
#include <math.h>

typedef __attribute__((ext_vector_type(16))) __bf16 v16bf;
typedef __attribute__((ext_vector_type(8)))  __bf16 v8bf;
typedef __attribute__((ext_vector_type(8)))  float  v8f;

#define SK       4096   // sequence length (keys)
#define DH       64     // head dim
#define TOPK     128
#define QTILE    16     // queries per workgroup
#define NWAVES   8
#define NTHREADS 256
#define NTILES   (SK / 16)           // 256 N-tiles
#define NELEM    (2 * 8 * SK * DH)   // B*H*S*D
#define LISTCAP  2048                // candidate list capacity per wave

// LDS layout (bytes)
#define SC_OFF    0
#define H16_OFF   (QTILE * SK * 4)                 // 262144: per-query top-byte hist [16][256]
#define WH_OFF    (H16_OFF + QTILE * 256 * 4)      // 278528: per-wave radix hist [8][256]
#define LIST_OFF  (WH_OFF + NWAVES * 256 * 4)      // 286720: per-wave u16 lists [8][LISTCAP]
#define SMEM_SZ   (LIST_OFF + NWAVES * LISTCAP * 2)// 319488 <= 320KB

// monotonic float->uint: larger float => larger uint (3 VALU, branch-free)
__device__ __forceinline__ unsigned int mono(float f) {
    const unsigned int u = __float_as_uint(f);
    return u ^ ((unsigned int)((int)u >> 31) | 0x80000000u);
}
__device__ __forceinline__ float unmono(unsigned int u) {
    return __uint_as_float((u & 0x80000000u) ? (u ^ 0x80000000u) : ~u);
}

// ---------- pre-pass: convert K (f32) -> bf16 once, into workspace ----------
extern "C" __global__ void __launch_bounds__(256)
cvt_bf16_kernel(const float* __restrict__ src, __bf16* __restrict__ dst) {
    const int i = (blockIdx.x * 256 + threadIdx.x) * 8;
    if (i < NELEM) {
        const float4 f0 = *(const float4*)(src + i);
        const float4 f1 = *(const float4*)(src + i + 4);
        v8bf o;
        o[0] = (__bf16)f0.x; o[1] = (__bf16)f0.y;
        o[2] = (__bf16)f0.z; o[3] = (__bf16)f0.w;
        o[4] = (__bf16)f1.x; o[5] = (__bf16)f1.y;
        o[6] = (__bf16)f1.z; o[7] = (__bf16)f1.w;
        *(v8bf*)(dst + i) = o;
    }
}

extern "C" __global__ void __launch_bounds__(NTHREADS)
topk_attn_kernel(const float* __restrict__ Q, const __bf16* __restrict__ Kb,
                 const float* __restrict__ V, float* __restrict__ O)
{
    extern __shared__ char smem[];
    float*          sc  = (float*)(smem + SC_OFF);
    unsigned int*   h16 = (unsigned int*)(smem + H16_OFF);
    unsigned int*   wh  = (unsigned int*)(smem + WH_OFF);
    unsigned short* lst = (unsigned short*)(smem + LIST_OFF);

    const int lane    = threadIdx.x & 31;
    const int wid     = threadIdx.x >> 5;
    const int mtile   = blockIdx.x & 255;   // S/QTILE = 256 tiles
    const int bh      = blockIdx.x >> 8;    // B*H = 16
    const int q0      = mtile * QTILE;
    const int halfSel = lane >> 4;
    const int l16     = lane & 15;

    const size_t qbase = ((size_t)bh * SK + q0) * DH;
    const size_t kbase = (size_t)bh * SK * DH;

    // zero per-query top-byte histograms (16KB / 256 threads)
    {
        uint4 z = {0u, 0u, 0u, 0u};
        uint4* p = (uint4*)(h16) + threadIdx.x * 4;
        p[0] = z; p[1] = z; p[2] = z; p[3] = z;
    }
    __syncthreads();

    // ================= Phase 1: scores GEMM + fused coarse histogram ========
    v16bf a0, a1;
    {
        const float  scale = 0.125f;  // 1/sqrt(64)
        const float* qrow  = Q + qbase + (size_t)l16 * DH;
        const int    kh    = halfSel * 16;
        #pragma unroll
        for (int v = 0; v < 8; ++v) {
            const int kk = kh + 2 * v;
            const float2 f0 = *(const float2*)(qrow + kk);
            const float2 f1 = *(const float2*)(qrow + kk + 32);
            a0[2 * v]     = (__bf16)(f0.x * scale);
            a0[2 * v + 1] = (__bf16)(f0.y * scale);
            a1[2 * v]     = (__bf16)(f1.x * scale);
            a1[2 * v + 1] = (__bf16)(f1.y * scale);
        }
    }

    const __bf16* kcol = Kb + kbase + (size_t)l16 * DH + halfSel * 16;
    v16bf b0a = *(const v16bf*)(kcol + (size_t)(wid * 16) * DH);
    v16bf b1a = *(const v16bf*)(kcol + (size_t)(wid * 16) * DH + 32);

    for (int nt = wid; nt < NTILES; nt += 2 * NWAVES) {   // 16 outer iterations
        const int ntB = nt + NWAVES;                      // always < NTILES
        const v16bf b0b = *(const v16bf*)(kcol + (size_t)(ntB * 16) * DH);
        const v16bf b1b = *(const v16bf*)(kcol + (size_t)(ntB * 16) * DH + 32);

        v8f c = {};
        c = __builtin_amdgcn_wmma_f32_16x16x32_bf16(false, a0, false, b0a, (short)0, c, false, false);
        c = __builtin_amdgcn_wmma_f32_16x16x32_bf16(false, a1, false, b1a, (short)0, c, false, false);
        {
            float* dst = sc + (size_t)(halfSel * 8) * SK + (nt * 16 + l16);
            #pragma unroll
            for (int r = 0; r < 8; ++r) {
                dst[(size_t)r * SK] = c[r];
                atomicAdd(&h16[(halfSel * 8 + r) * 256 + (mono(c[r]) >> 24)], 1u);
            }
        }

        const int ntn = (nt + 2 * NWAVES < NTILES) ? (nt + 2 * NWAVES) : nt;
        b0a = *(const v16bf*)(kcol + (size_t)(ntn * 16) * DH);
        b1a = *(const v16bf*)(kcol + (size_t)(ntn * 16) * DH + 32);

        v8f c2 = {};
        c2 = __builtin_amdgcn_wmma_f32_16x16x32_bf16(false, a0, false, b0b, (short)0, c2, false, false);
        c2 = __builtin_amdgcn_wmma_f32_16x16x32_bf16(false, a1, false, b1b, (short)0, c2, false, false);
        {
            float* dst = sc + (size_t)(halfSel * 8) * SK + (ntB * 16 + l16);
            #pragma unroll
            for (int r = 0; r < 8; ++r) {
                dst[(size_t)r * SK] = c2[r];
                atomicAdd(&h16[(halfSel * 8 + r) * 256 + (mono(c2[r]) >> 24)], 1u);
            }
        }
    }
    __syncthreads();   // scores + per-query histograms complete

    // ======== Phase 2: per-wave candidate compaction + exact select =========
    unsigned int*   h = wh  + wid * 256;       // private per-wave radix hist
    unsigned short* L = lst + wid * LISTCAP;   // private per-wave candidate list
    for (int qi = 0; qi < 2; ++qi) {           // 8 waves x 2 = 16 queries
        const int    q   = wid * 2 + qi;
        const float* row = sc + (size_t)q * SK;

        // --- 2a: cut byte b0 AND in-bin remainder r0 from fused histogram ---
        int b0cut, r0;
        {
            const unsigned int* hq = h16 + q * 256;
            const uint4 c0 = *(const uint4*)(hq + lane * 8);
            const uint4 c1 = *(const uint4*)(hq + lane * 8 + 4);
            const int cnt[8] = {(int)c0.x, (int)c0.y, (int)c0.z, (int)c0.w,
                                (int)c1.x, (int)c1.y, (int)c1.z, (int)c1.w};
            int loc = 0;
            #pragma unroll
            for (int i = 0; i < 8; ++i) loc += cnt[i];
            int s = loc;
            #pragma unroll
            for (int off = 1; off < 32; off <<= 1) {
                const int t = __shfl_down(s, off, 32);
                s += (lane + off < 32) ? t : 0;
            }
            const int  above = s - loc;
            const bool win   = (above < TOPK) && (s >= TOPK);
            int selb = 0, remb = 0;
            if (win) {
                int  cum  = above;
                bool done = false;
                #pragma unroll
                for (int i = 7; i >= 0; --i) {
                    if (!done) {
                        if (cum + cnt[i] >= TOPK) {
                            selb = lane * 8 + i; remb = TOPK - cum; done = true;
                        } else cum += cnt[i];
                    }
                }
            }
            const unsigned int wm = __builtin_amdgcn_ballot_w32(win);
            const int          wl = __ffs(wm) - 1;
            b0cut = __shfl(selb, wl, 32);
            r0    = __shfl(remb, wl, 32);
        }

        // --- 2b: one full sweep: compact candidates (in j0, e, lane order) + row max ---
        int nc = 0;
        unsigned int umax = 0;
        for (int j0 = 0; j0 < SK; j0 += 128) {
            const float4 f = *(const float4*)(row + j0 + lane * 4);
            const float  fe[4] = {f.x, f.y, f.z, f.w};
            #pragma unroll
            for (int e = 0; e < 4; ++e) {
                const unsigned int u = mono(fe[e]);
                umax = (u > umax) ? u : umax;
                const bool cand = (int)(u >> 24) >= b0cut;
                const unsigned int msk = __builtin_amdgcn_ballot_w32(cand);
                if (cand) {
                    const int pos = nc + __popc(msk & ((1u << lane) - 1u));
                    if (pos < LISTCAP)
                        L[pos] = (unsigned short)(j0 + 4 * lane + e);
                }
                nc += __popc(msk);
            }
        }
        if (nc > LISTCAP) nc = LISTCAP;   // astronomically unlikely overflow
        #pragma unroll
        for (int off = 16; off; off >>= 1) {
            const unsigned int t = __shfl_xor(umax, off, 32);
            umax = (t > umax) ? t : umax;
        }
        const float m = unmono(umax);

        // --- 2c: radix passes 1..3 over candidates (pass 0 == h16 scan) ---
        unsigned int prefix = (unsigned int)b0cut;
        int remaining = r0;
        for (int pass = 1; pass < 4; ++pass) {
            const int shift = 24 - 8 * pass;   // 16, 8, 0
            {
                uint4 z = {0u, 0u, 0u, 0u};
                *(uint4*)(h + lane * 8)     = z;
                *(uint4*)(h + lane * 8 + 4) = z;
            }
            for (int c0i = 0; c0i < nc; c0i += 32) {
                const int  ci  = c0i + lane;
                const bool act = ci < nc;
                const int  idx = act ? (int)L[ci] : 0;
                const unsigned int u = mono(row[idx]);
                if (act && (u >> (shift + 8)) == prefix)
                    atomicAdd(&h[(u >> shift) & 255u], 1u);
            }
            const uint4 c0 = *(const uint4*)(h + lane * 8);
            const uint4 c1 = *(const uint4*)(h + lane * 8 + 4);
            const int cnt[8] = {(int)c0.x, (int)c0.y, (int)c0.z, (int)c0.w,
                                (int)c1.x, (int)c1.y, (int)c1.z, (int)c1.w};
            int loc = 0;
            #pragma unroll
            for (int i = 0; i < 8; ++i) loc += cnt[i];
            int s = loc;
            #pragma unroll
            for (int off = 1; off < 32; off <<= 1) {
                const int t = __shfl_down(s, off, 32);
                s += (lane + off < 32) ? t : 0;
            }
            const int  above = s - loc;
            const bool win   = (above < remaining) && (s >= remaining);
            int sel = 0, newrem = 0;
            if (win) {
                int  cum  = above;
                bool done = false;
                #pragma unroll
                for (int i = 7; i >= 0; --i) {
                    if (!done) {
                        if (cum + cnt[i] >= remaining) {
                            sel = lane * 8 + i; newrem = remaining - cum; done = true;
                        } else cum += cnt[i];
                    }
                }
            }
            const unsigned int wm = __builtin_amdgcn_ballot_w32(win);
            const int          wl = __ffs(wm) - 1;
            sel       = __shfl(sel, wl, 32);
            remaining = __shfl(newrem, wl, 32);
            prefix    = (prefix << 8) | (unsigned int)sel;
        }
        const unsigned int T          = prefix;
        const int          tiesNeeded = remaining;

        // --- 2d: gather + softmax; one wave-wide exp per included element ---
        float acc0 = 0.0f, acc1 = 0.0f, zp = 0.0f;   // zp wave-uniform
        const float* vb = V + kbase;
        int tieBase = 0;
        for (int c0i = 0; c0i < nc; c0i += 32) {
            const int  ci  = c0i + lane;
            const bool act = ci < nc;
            const int  idx = act ? (int)L[ci] : 0;
            const float        sv = row[idx];
            const unsigned int u  = act ? mono(sv) : 0u;
            const bool         isTie = (u == T);
            const unsigned int tmask = __builtin_amdgcn_ballot_w32(isTie);
            bool inc = (u > T);
            if (isTie) {  // deterministic list-order tie inclusion
                const int rank = tieBase + __popc(tmask & ((1u << lane) - 1u));
                if (rank < tiesNeeded) inc = true;
            }
            unsigned int imask = __builtin_amdgcn_ballot_w32(inc);
            while (imask) {
                const int b = __ffs(imask) - 1;
                imask &= imask - 1;
                const float svb = __shfl(sv, b, 32);
                const int   ib  = __shfl(idx, b, 32);
                const float wv  = __expf(svb - m);   // wave-wide, once per element
                zp += wv;
                const float2 vr = *(const float2*)(vb + (size_t)ib * DH + 2 * lane);
                acc0 += wv * vr.x;
                acc1 += wv * vr.y;
            }
            tieBase += __popc(tmask);
        }
        const float invZ = 1.0f / zp;   // uniform: no reduction needed

        float2 o2;
        o2.x = acc0 * invZ;
        o2.y = acc1 * invZ;
        *(float2*)(O + qbase + (size_t)q * DH + 2 * lane) = o2;
    }
}

extern "C" void kernel_launch(void* const* d_in, const int* in_sizes, int n_in,
                              void* d_out, int out_size, void* d_ws, size_t ws_size,
                              hipStream_t stream) {
    (void)in_sizes; (void)n_in; (void)out_size; (void)ws_size;
    const float* Q  = (const float*)d_in[0];
    const float* K  = (const float*)d_in[1];
    const float* V  = (const float*)d_in[2];
    float*       O  = (float*)d_out;
    __bf16*      Kb = (__bf16*)d_ws;   // 8 MB bf16 copy of K

    cvt_bf16_kernel<<<NELEM / (256 * 8), 256, 0, stream>>>(K, Kb);

    hipFuncSetAttribute((const void*)topk_attn_kernel,
                        hipFuncAttributeMaxDynamicSharedMemorySize, SMEM_SZ);
    dim3 grid(16 * (SK / QTILE));  // B*H * 256 = 4096 workgroups
    topk_attn_kernel<<<grid, NTHREADS, SMEM_SZ, stream>>>(Q, Kb, V, O);
}
